// _PointnetSAModuleBase_84232898609384
// MI455X (gfx1250) — matrix operand order repaired and checked by reading, not compiled, not run on hardware
//
#include <hip/hip_runtime.h>

// ---------------- problem constants (from reference) ----------------
#define B_   8
#define N_   8192
#define C_   64
#define S_   1024      // NPOINT
#define K_   64        // NSAMPLE
#define CIN  67        // 3 + C
#define KP1  96        // CIN padded up to multiple of 32 for WMMA k-steps
#define RAD2 (0.2f * 0.2f)

// f16 weight workspace layout (element offsets)
#define W2OFF  (KP1 * 64)                       // 6144
#define W3OFF  (KP1 * 64 + 64 * 64)             // 10240
#define WTOTAL (KP1 * 64 + 64 * 64 + 64 * 128)  // 18432 halves = 36 KB

typedef __attribute__((ext_vector_type(16))) _Float16 v16h;
typedef __attribute__((ext_vector_type(8)))  float    v8f;
typedef int v4i __attribute__((vector_size(4 * sizeof(int))));
typedef __attribute__((address_space(1))) v4i* g_v4i_ptr;   // global int4*
typedef __attribute__((address_space(3))) v4i* l_v4i_ptr;   // LDS int4*

// =====================================================================
// Kernel 0: one-shot weight conversion f32 -> f16 into d_ws, with W1's
// K padded 67 -> 96 with zeros so layer 1 is exactly 3 WMMA k-steps.
// =====================================================================
__global__ __launch_bounds__(256) void wprep_kernel(const float* __restrict__ W1,
                                                    const float* __restrict__ W2,
                                                    const float* __restrict__ W3,
                                                    _Float16* __restrict__ wsf16) {
  const int e = blockIdx.x * 256 + threadIdx.x;
  if (e >= WTOTAL) return;
  float v;
  if (e < W2OFF) {
    const int k = e / 64;                 // W1 is (67,64) row-major, pad rows 67..95
    v = (k < CIN) ? W1[e] : 0.0f;
  } else if (e < W3OFF) {
    v = W2[e - W2OFF];
  } else {
    v = W3[e - W3OFF];
  }
  wsf16[e] = (_Float16)v;
}

// =====================================================================
// Kernel 1: furthest point sampling, one block per batch cloud.
// 1024 threads, each owns 8 points in registers. Per pick: local min-update
// + argmax, wave32 shuffle reduce, 32-partial LDS reduce. Tie-break = lowest
// index (matches jnp.argmax).
// =====================================================================
__global__ __launch_bounds__(1024) void fps_kernel(const float* __restrict__ xyz,
                                                   float* __restrict__ out_xyz) {
  __shared__ float slast[3];
  __shared__ float sval[32];
  __shared__ int   sidx[32];
  const int b = blockIdx.x;
  const int t = threadIdx.x;
  const float* X = xyz + (size_t)b * N_ * 3;

  float px[8], py[8], pz[8], d[8];
#pragma unroll
  for (int i = 0; i < 8; ++i) {
    int n = i * 1024 + t;
    px[i] = X[n * 3 + 0];
    py[i] = X[n * 3 + 1];
    pz[i] = X[n * 3 + 2];
    d[i] = 1e10f;
  }
  if (t == 0) {
    slast[0] = X[0]; slast[1] = X[1]; slast[2] = X[2];
    float* o = out_xyz + (size_t)b * S_ * 3;
    o[0] = X[0]; o[1] = X[1]; o[2] = X[2];   // first pick is index 0
  }
  __syncthreads();

  for (int it = 1; it < S_; ++it) {
    const float lx = slast[0], ly = slast[1], lz = slast[2];
    float bestv = -1.0f;
    int   besti = 0x7fffffff;
#pragma unroll
    for (int i = 0; i < 8; ++i) {
      float dx = px[i] - lx, dy = py[i] - ly, dz = pz[i] - lz;
      float dd = dx * dx + dy * dy + dz * dz;
      d[i] = fminf(d[i], dd);
      int n = i * 1024 + t;
      if (d[i] > bestv || (d[i] == bestv && n < besti)) { bestv = d[i]; besti = n; }
    }
#pragma unroll
    for (int off = 16; off > 0; off >>= 1) {
      float ov = __shfl_down(bestv, off, 32);
      int   oi = __shfl_down(besti, off, 32);
      if (ov > bestv || (ov == bestv && oi < besti)) { bestv = ov; besti = oi; }
    }
    __syncthreads();
    if ((t & 31) == 0) { sval[t >> 5] = bestv; sidx[t >> 5] = besti; }
    __syncthreads();
    if (t < 32) {
      bestv = sval[t]; besti = sidx[t];
#pragma unroll
      for (int off = 16; off > 0; off >>= 1) {
        float ov = __shfl_down(bestv, off, 32);
        int   oi = __shfl_down(besti, off, 32);
        if (ov > bestv || (ov == bestv && oi < besti)) { bestv = ov; besti = oi; }
      }
      if (t == 0) {
        float nx = X[besti * 3 + 0], ny = X[besti * 3 + 1], nz = X[besti * 3 + 2];
        slast[0] = nx; slast[1] = ny; slast[2] = nz;
        float* o = out_xyz + ((size_t)b * S_ + it) * 3;
        o[0] = nx; o[1] = ny; o[2] = nz;
      }
    }
    __syncthreads();
  }
}

// =====================================================================
// WMMA fragment helpers (layouts per CDNA5 ISA 7.12.2, wave32)
// =====================================================================

// A fragment: 16x32 f16 tile, rows m0..m0+15, cols k0..k0+31 of A[lda].
// lanes 0-15  (M=lane):    halves 0..7 -> K=k0+0..7,  halves 8..15 -> K=k0+16..23
// lanes 16-31 (M=lane-16): halves 0..7 -> K=k0+8..15, halves 8..15 -> K=k0+24..31
__device__ inline v16h load_a_frag(const _Float16* A, int lda, int m0, int k0, int lane) {
  const int hs = (lane >> 4) & 1;
  const _Float16* row = A + (m0 + (lane & 15)) * lda + k0 + hs * 8;
  v16h a;
#pragma unroll
  for (int i = 0; i < 8; ++i) a[i] = row[i];
#pragma unroll
  for (int i = 0; i < 8; ++i) a[8 + i] = row[16 + i];
  return a;
}

// B fragment: 32x16 f16 tile, rows k0..k0+31, cols n0..n0+15 of B[ldb].
// lanes 0-15  (N=lane):    halves 0..7 -> K=k0+0..7,  halves 8..15 -> K=k0+16..23
// lanes 16-31 (N=lane-16): halves 0..7 -> K=k0+8..15, halves 8..15 -> K=k0+24..31
__device__ inline v16h load_b_frag(const _Float16* Bm, int ldb, int k0, int n0, int lane) {
  const int hs = (lane >> 4) & 1;
  const _Float16* col = Bm + (k0 + hs * 8) * ldb + n0 + (lane & 15);
  v16h bfr;
#pragma unroll
  for (int i = 0; i < 16; ++i) bfr[i] = col[(i < 8 ? i : i + 8) * ldb];
  return bfr;
}

// One 16x16 output tile, accumulating ktiles k-steps of 32.
__device__ inline v8f mm_tile(const _Float16* A, int lda, const _Float16* Bm, int ldb,
                              int mt, int nt, int ktiles, int lane) {
  v8f c = {};
#pragma unroll
  for (int kt = 0; kt < ktiles; ++kt) {
    v16h a   = load_a_frag(A, lda, mt * 16, kt * 32, lane);
    v16h bfr = load_b_frag(Bm, ldb, kt * 32, nt * 16, lane);
    c = __builtin_amdgcn_wmma_f32_16x16x32_f16(false, a, false, bfr,
                                               (short)0, c, false, false);
  }
  return c;
}

// C/D layout: VGPR r holds (M = mt*16 + r + 8*(lane>=16), N = nt*16 + (lane&15))
__device__ inline void store_tile_relu(v8f c, _Float16* H, int ldh, const float* bias,
                                       int mt, int nt, int lane) {
  const int n  = nt * 16 + (lane & 15);
  const int mb = mt * 16 + ((lane >> 4) & 1) * 8;
  const float bv = bias[n];
#pragma unroll
  for (int r = 0; r < 8; ++r) {
    float v = c[r] + bv;
    v = v > 0.0f ? v : 0.0f;
    H[(mb + r) * ldh + n] = (_Float16)v;
  }
}

// Layer-3 tile: bias+relu then max over the 8 M rows this lane holds; values
// are >= 0 after ReLU so int-bit atomicMax on LDS is order-correct.
__device__ inline void pool_tile(v8f c, const float* bias, float* pool, int nt, int lane) {
  const int n = nt * 16 + (lane & 15);
  const float bv = bias[n];
  float mx = 0.0f;
#pragma unroll
  for (int r = 0; r < 8; ++r) {
    float v = c[r] + bv;
    v = v > 0.0f ? v : 0.0f;
    mx = fmaxf(mx, v);
  }
  atomicMax((int*)&pool[n], __float_as_int(mx));
}

// Stage `halves` f16 weights (16B-aligned, multiple of 8) from global -> LDS.
// Prefers the gfx1250 async-copy path (ASYNCcnt); falls back to b128 copies.
__device__ inline void stage_weights(const _Float16* __restrict__ src,
                                     _Float16* dst, int halves, int tid) {
#if __has_builtin(__builtin_amdgcn_global_load_async_to_lds_b128)
  for (int e = tid * 8; e < halves; e += 256 * 8) {
    __builtin_amdgcn_global_load_async_to_lds_b128(
        (g_v4i_ptr)(src + e),           // global source (AS1 int4*)
        (l_v4i_ptr)(dst + e),           // LDS destination (AS3 int4*)
        /*imm offset*/ 0, /*cpol*/ 0);
  }
#if __has_builtin(__builtin_amdgcn_s_wait_asynccnt)
  __builtin_amdgcn_s_wait_asynccnt(0);
#else
  asm volatile("s_wait_asynccnt 0x0" ::: "memory");
#endif
#else
  for (int e = tid * 8; e < halves; e += 256 * 8) {
    *(float4*)(dst + e) = *(const float4*)(src + e);
  }
#endif
}

// =====================================================================
// Kernel 2: one block (256 threads = 8 waves) per center.
// ball query (ordered, wave 0) -> gather f16 tile -> 3x WMMA GEMM layers
// (64x96 @ 96x64, 64x64 @ 64x64, 64x64 @ 64x128) -> max-pool -> output.
// =====================================================================
__global__ __launch_bounds__(256) void group_mlp_kernel(
    const float* __restrict__ xyz, const float* __restrict__ feat,
    const _Float16* __restrict__ Wp,          // prepped f16 weights in d_ws
    const float* __restrict__ b1, const float* __restrict__ b2,
    const float* __restrict__ b3,
    const float* __restrict__ new_xyz, float* __restrict__ out_feat) {
  __shared__ int      nbr[K_];
  __shared__ int      s_cnt;
  __shared__ float    s_ctr[3];
  __shared__ __align__(16) _Float16 Ag[K_ * KP1];    // 64 x 96 grouped input (f16)
  __shared__ __align__(16) _Float16 wbuf[64 * 128];  // per-layer weight staging (16 KB)
  __shared__ __align__(16) _Float16 H1[K_ * 64];
  __shared__ __align__(16) _Float16 H2[K_ * 64];
  __shared__ float    b1f[64], b2f[64], b3f[128];
  __shared__ float    pool[128];

  const int tid  = threadIdx.x;
  const int lane = tid & 31;
  const int wave = tid >> 5;
  const int bs   = blockIdx.x;          // b * S_ + s
  const int b    = bs / S_;
  const int s    = bs % S_;
  const float* X = xyz  + (size_t)b * N_ * 3;
  const float* F = feat + (size_t)b * C_ * N_;

  if (tid == 0) s_cnt = 0;
  if (tid < 3)  s_ctr[tid] = new_xyz[(size_t)bs * 3 + tid];
  if (tid < 64) { b1f[tid] = b1[tid]; b2f[tid] = b2[tid]; }
  if (tid < 128) { b3f[tid] = b3[tid]; pool[tid] = 0.0f; }
  __syncthreads();

  // ---- ordered ball query: first K_ points within radius, index order ----
  if (wave == 0) {
    const float cx = s_ctr[0], cy = s_ctr[1], cz = s_ctr[2];
    int cnt = 0;
    for (int base = 0; base < N_ && cnt < K_; base += 32) {
      const int n = base + lane;
      if (base + 256 < N_)                       // stream prefetch ~8 iters ahead
        __builtin_prefetch(X + (base + 256 + lane) * 3, 0, 1);
      const float dx = X[n * 3 + 0] - cx;
      const float dy = X[n * 3 + 1] - cy;
      const float dz = X[n * 3 + 2] - cz;
      const bool valid = (dx * dx + dy * dy + dz * dz) < RAD2;
      const unsigned mask = (unsigned)__ballot(valid);   // wave32: low 32 bits
      const int rank = __popc(mask & ((1u << lane) - 1u));
      if (valid && (cnt + rank) < K_) nbr[cnt + rank] = n;
      cnt += __popc(mask);
    }
    if (lane == 0) s_cnt = (cnt > K_) ? K_ : cnt;
  }
  __syncthreads();
  {
    const int cnt = s_cnt;
    int v = 0;                                   // empty ball -> index 0
    if (tid < K_ && cnt > 0) v = (tid < cnt) ? nbr[tid] : nbr[0];  // pad w/ first
    __syncthreads();
    if (tid < K_) nbr[tid] = v;
  }
  __syncthreads();

  // ---- gather grouped input (relative xyz ++ features) + stage W1 ----
  {
    const float cx = s_ctr[0], cy = s_ctr[1], cz = s_ctr[2];
    for (int e = tid; e < K_ * KP1; e += 256) {
      const int j = e / KP1, col = e % KP1;
      const int p = nbr[j];
      float v;
      if (col < 3)        v = X[p * 3 + col] - (col == 0 ? cx : (col == 1 ? cy : cz));
      else if (col < CIN) v = F[(size_t)(col - 3) * N_ + p];
      else                v = 0.0f;
      Ag[e] = (_Float16)v;
    }
    stage_weights(Wp, wbuf, KP1 * 64, tid);      // W1 padded (96 x 64)
  }
  __syncthreads();

  // ---- layer 1: H1 = relu(Ag(64x96) @ W1(96x64) + b1) : tiles t = wave, wave+8
#pragma unroll
  for (int r = 0; r < 2; ++r) {
    const int t = wave + r * 8;
    const int mt = t >> 2, nt = t & 3;
    v8f c = mm_tile(Ag, KP1, wbuf, 64, mt, nt, 3, lane);
    store_tile_relu(c, H1, 64, b1f, mt, nt, lane);
  }
  __syncthreads();

  stage_weights(Wp + W2OFF, wbuf, 64 * 64, tid);
  __syncthreads();

  // ---- layer 2: H2 = relu(H1 @ W2(64x64) + b2) ----
#pragma unroll
  for (int r = 0; r < 2; ++r) {
    const int t = wave + r * 8;
    const int mt = t >> 2, nt = t & 3;
    v8f c = mm_tile(H1, 64, wbuf, 64, mt, nt, 2, lane);
    store_tile_relu(c, H2, 64, b2f, mt, nt, lane);
  }
  __syncthreads();

  stage_weights(Wp + W3OFF, wbuf, 64 * 128, tid);
  __syncthreads();

  // ---- layer 3 + max-pool over neighbors: relu(H2 @ W3(64x128) + b3) ----
#pragma unroll
  for (int r = 0; r < 4; ++r) {
    const int t = wave + r * 8;
    const int mt = t >> 3, nt = t & 7;
    v8f c = mm_tile(H2, 64, wbuf, 128, mt, nt, 2, lane);
    pool_tile(c, b3f, pool, nt, lane);
  }
  __syncthreads();

  // output layout (B, 128, S)
  if (tid < 128) out_feat[((size_t)b * 128 + tid) * S_ + s] = pool[tid];
}

// =====================================================================
extern "C" void kernel_launch(void* const* d_in, const int* in_sizes, int n_in,
                              void* d_out, int out_size, void* d_ws, size_t ws_size,
                              hipStream_t stream) {
  (void)in_sizes; (void)n_in; (void)out_size; (void)ws_size;
  const float* xyz  = (const float*)d_in[0];
  const float* feat = (const float*)d_in[1];
  const float* W1   = (const float*)d_in[2];
  const float* b1   = (const float*)d_in[3];
  const float* W2   = (const float*)d_in[4];
  const float* b2   = (const float*)d_in[5];
  const float* W3   = (const float*)d_in[6];
  const float* b3   = (const float*)d_in[7];

  float* out      = (float*)d_out;
  float* out_xyz  = out;                       // (B, 1024, 3)
  float* out_feat = out + (size_t)B_ * S_ * 3; // (B, 128, 1024)
  _Float16* wsf16 = (_Float16*)d_ws;           // 36 KB of f16 weights

  wprep_kernel<<<(WTOTAL + 255) / 256, 256, 0, stream>>>(W1, W2, W3, wsf16);
  fps_kernel<<<B_, 1024, 0, stream>>>(xyz, out_xyz);
  group_mlp_kernel<<<B_ * S_, 256, 0, stream>>>(xyz, feat, wsf16, b1, b2, b3,
                                                out_xyz, out_feat);
}
